// MoEBertSelfAttention_38534446579845
// MI455X (gfx1250) — matrix operand order repaired and checked by reading, not compiled, not run on hardware
//
#include <hip/hip_runtime.h>
#include <hip/hip_bf16.h>

// ---------------------------------------------------------------------------
// MoE BERT self-attention for gfx1250 (MI455X), f16 WMMA path, fp32 accum.
// B=16, S=512, H=768, NH=12, DH=64, E=8
// ---------------------------------------------------------------------------

#define NB   16
#define NS   512
#define NH_  768
#define NHEAD 12
#define DH_  64
#define NE   8

typedef _Float16 v16h __attribute__((ext_vector_type(16)));
typedef _Float16 v8h  __attribute__((ext_vector_type(8)));
typedef _Float16 v4h  __attribute__((ext_vector_type(4)));
typedef float    v8f  __attribute__((ext_vector_type(8)));
typedef int      v4i  __attribute__((ext_vector_type(4)));

typedef __attribute__((address_space(1))) v4i* glb_v4i_ptr;
typedef __attribute__((address_space(3))) v4i* lds_v4i_ptr;

#ifndef __has_builtin
#define __has_builtin(x) 0
#endif

#if __has_builtin(__builtin_amdgcn_global_load_async_to_lds_b128)
#define HAVE_ASYNC_LDS 1
#else
#define HAVE_ASYNC_LDS 0
#endif

#if __has_builtin(__builtin_amdgcn_s_wait_asynccnt)
#define WAIT_ASYNCCNT(n) __builtin_amdgcn_s_wait_asynccnt(n)
#else
#define WAIT_ASYNCCNT(n) asm volatile("s_wait_asynccnt %0" :: "n"(n) : "memory")
#endif

static __device__ __forceinline__ v8h ld8(const _Float16* p) {
    return *(const v8h*)p;
}
static __device__ __forceinline__ v16h mk16(v8h lo, v8h hi) {
    return __builtin_shufflevector(lo, hi, 0,1,2,3,4,5,6,7,8,9,10,11,12,13,14,15);
}
static __device__ __forceinline__ v8f wmma16(v16h a, v16h b, v8f c) {
    // (neg_a, A, neg_b, B, c_mod, C, reuse_a, reuse_b)
    return __builtin_amdgcn_wmma_f32_16x16x32_f16(false, a, false, b, (short)0, c, false, false);
}

// 16-byte global -> LDS copy, async when the gfx1250 builtin is available.
// Builtin signature (from hipcc diagnostic): (v4i AS1*, v4i AS3*, imm, imm).
// Flat LDS pointers: low 32 bits are the LDS byte offset (ISA §10.2 aperture),
// so integer-casting to AS3 yields a valid LDS address.
static __device__ __forceinline__ void async_cp16(const _Float16* g, _Float16* l) {
#if HAVE_ASYNC_LDS
    __builtin_amdgcn_global_load_async_to_lds_b128(
        (glb_v4i_ptr)(uintptr_t)g,
        (lds_v4i_ptr)(uintptr_t)l, 0, 0);
#else
    *(v8h*)l = *(const v8h*)g;
#endif
}

// ---------------------------------------------------------------------------
// fp32 -> f16 conversion (4-wide)
// ---------------------------------------------------------------------------
__global__ void cvt_f32_to_f16_4(const float4* __restrict__ src,
                                 v4h* __restrict__ dst, int n4) {
    int i = blockIdx.x * blockDim.x + threadIdx.x;
    if (i < n4) {
        float4 f = src[i];
        v4h h;
        h[0] = (_Float16)f.x; h[1] = (_Float16)f.y;
        h[2] = (_Float16)f.z; h[3] = (_Float16)f.w;
        dst[i] = h;
    }
}

// ---------------------------------------------------------------------------
// QKV projection: per (batch b, proj p) GEMM  Y[s,o] = sum_d X[s,d]*W[o,d] + bias[o]
// grid = (S/128, H/128, B*3), block = 256 (8 waves, 2x4, each 64x32)
// Double-buffered async-to-LDS staging of the 128x32 X and W tiles.
// Q,K stored [B,NH,S,DH] f16 ; V stored transposed [B,NH,DH,S] f16
// ---------------------------------------------------------------------------
__global__ __launch_bounds__(256)
void qkv_wmma_kernel(const _Float16* __restrict__ Xh,
                     const _Float16* __restrict__ Wall,   // [3][E][H][H] f16
                     const float* __restrict__ bq,
                     const float* __restrict__ bk,
                     const float* __restrict__ bv,
                     const int* __restrict__ eidx,
                     _Float16* __restrict__ Qh,
                     _Float16* __restrict__ Kh,
                     _Float16* __restrict__ Vt) {
    const int z = blockIdx.z;
    const int b = z / 3;
    const int p = z % 3;
    const int e = eidx[b];

    const int m0blk = blockIdx.x * 128;
    const int n0blk = blockIdx.y * 128;

    // block-local global tile bases
    const _Float16* Xg = Xh + (size_t)b * NS * NH_ + (size_t)m0blk * NH_;
    const _Float16* Wg = Wall + ((size_t)p * NE + e) * NH_ * NH_ + (size_t)n0blk * NH_;
    const float* bias = (p == 0 ? bq : (p == 1 ? bk : bv)) + (size_t)e * NH_;

    const int t      = threadIdx.x;
    const int lane   = t & 31;
    const int wave   = t >> 5;
    const int wm     = wave >> 2;       // 0..1
    const int wn     = wave & 3;        // 0..3
    const int laneM  = lane & 15;
    const int laneHi = lane >> 4;

    // [2 bufs][128 rows][32 halfs]  (8 KB per tile per buffer, 32 KB total)
    __shared__ __align__(16) _Float16 XsBuf[2][128 * 32];
    __shared__ __align__(16) _Float16 WsBuf[2][128 * 32];

    v8f c[4][2] = {};

    // --- issue one K-step tile pair (4 async b128 ops per thread) ---
    auto issue_tiles = [&](int buf, int kk) {
#pragma unroll
        for (int c2 = 0; c2 < 2; ++c2) {
            const int chunk = t + c2 * 256;          // 0..511
            const int row   = chunk >> 2;            // 0..127
            const int off   = (chunk & 3) * 8;       // halfs: 0,8,16,24
            async_cp16(Xg + (size_t)row * NH_ + kk + off,
                       &XsBuf[buf][row * 32 + off]);
            async_cp16(Wg + (size_t)row * NH_ + kk + off,
                       &WsBuf[buf][row * 32 + off]);
        }
    };

    const int NKS = NH_ / 32;   // 24 K-steps
    issue_tiles(0, 0);

    for (int ks = 0; ks < NKS; ++ks) {
        if (ks + 1 < NKS) {
            issue_tiles((ks + 1) & 1, (ks + 1) * 32);
            WAIT_ASYNCCNT(4);      // in-order: leaves only the prefetch pending
        } else {
            WAIT_ASYNCCNT(0);
        }
        __syncthreads();

        const _Float16* Xs = XsBuf[ks & 1];
        const _Float16* Ws = WsBuf[ks & 1];

        v16h a[4];
#pragma unroll
        for (int mi = 0; mi < 4; ++mi) {
            const _Float16* px = Xs + (wm * 64 + mi * 16 + laneM) * 32 + laneHi * 8;
            a[mi] = mk16(*(const v8h*)px, *(const v8h*)(px + 16));
        }
        v16h bm[2];
#pragma unroll
        for (int ni = 0; ni < 2; ++ni) {
            const _Float16* pw = Ws + (wn * 32 + ni * 16 + laneM) * 32 + laneHi * 8;
            bm[ni] = mk16(*(const v8h*)pw, *(const v8h*)(pw + 16));
        }
#pragma unroll
        for (int mi = 0; mi < 4; ++mi)
#pragma unroll
            for (int ni = 0; ni < 2; ++ni)
                c[mi][ni] = wmma16(a[mi], bm[ni], c[mi][ni]);

        __syncthreads();   // protect the buffer we are about to overwrite
    }

    // epilogue: bias add + store f16
#pragma unroll
    for (int ni = 0; ni < 2; ++ni) {
        const int n  = n0blk + wn * 32 + ni * 16 + laneM;   // output channel o
        const float bn = bias[n];
        const int head = n >> 6;                 // o / 64
        const int dh   = n & 63;                 // o % 64
#pragma unroll
        for (int mi = 0; mi < 4; ++mi) {
#pragma unroll
            for (int j = 0; j < 8; ++j) {
                const int m = m0blk + wm * 64 + mi * 16 + j + laneHi * 8;
                const float val = c[mi][ni][j] + bn;
                if (p == 0) {
                    Qh[(((size_t)b * NHEAD + head) * NS + m) * DH_ + dh] = (_Float16)val;
                } else if (p == 1) {
                    Kh[(((size_t)b * NHEAD + head) * NS + m) * DH_ + dh] = (_Float16)val;
                } else {
                    Vt[(((size_t)b * NHEAD + head) * DH_ + dh) * NS + m] = (_Float16)val;
                }
            }
        }
    }
}

// ---------------------------------------------------------------------------
// Attention: flash-style online softmax.
// grid = (S/64, B*NH), block = 128 (4 waves, each wave owns 16 q-rows)
// ---------------------------------------------------------------------------
__global__ __launch_bounds__(128)
void attn_wmma_kernel(const _Float16* __restrict__ Qh,
                      const _Float16* __restrict__ Kh,
                      const _Float16* __restrict__ Vt,
                      const float* __restrict__ mask,   // [B,1,1,S]
                      float* __restrict__ out) {
    const int bh = blockIdx.y;
    const int b  = bh / NHEAD;
    const int h  = bh % NHEAD;

    const int lane   = threadIdx.x & 31;
    const int wave   = threadIdx.x >> 5;
    const int laneM  = lane & 15;
    const int laneHi = lane >> 4;

    const int q0 = blockIdx.x * 64 + wave * 16;

    const size_t qkbase = ((size_t)b * NHEAD + h) * NS * DH_;   // [.., S, DH]
    const size_t vbase  = ((size_t)b * NHEAD + h) * DH_ * NS;   // [.., DH, S]

    // Q A-fragments (K-dim = head dim, two 32-steps), hoisted out of key loop
    v16h qa[2];
#pragma unroll
    for (int step = 0; step < 2; ++step) {
        const _Float16* pq = Qh + qkbase + (size_t)(q0 + laneM) * DH_
                           + step * 32 + laneHi * 8;
        qa[step] = mk16(ld8(pq), ld8(pq + 16));
    }

    float mrow[8], lrow[8];
#pragma unroll
    for (int j = 0; j < 8; ++j) { mrow[j] = -INFINITY; lrow[j] = 0.f; }
    v8f o[4] = {};

    __shared__ __align__(16) _Float16 pb[4][16][72];   // per-wave P tile, padded

    for (int kt = 0; kt < NS; kt += 64) {
        // ---- scores tile: 16 q-rows x 64 keys ----
        v8f sc[4] = {};
#pragma unroll
        for (int ni = 0; ni < 4; ++ni) {
#pragma unroll
            for (int step = 0; step < 2; ++step) {
                const int kn = kt + ni * 16 + laneM;
                const _Float16* pk = Kh + qkbase + (size_t)kn * DH_
                                   + step * 32 + laneHi * 8;
                v16h kb = mk16(ld8(pk), ld8(pk + 16));
                sc[ni] = wmma16(qa[step], kb, sc[ni]);
            }
        }

        // scale + additive mask (mask value is per-key => per C-fragment column)
        float mkv[4];
#pragma unroll
        for (int ni = 0; ni < 4; ++ni)
            mkv[ni] = mask[(size_t)b * NS + kt + ni * 16 + laneM];

        float tmax[8];
#pragma unroll
        for (int j = 0; j < 8; ++j) {
            float tm = -INFINITY;
#pragma unroll
            for (int ni = 0; ni < 4; ++ni) {
                float v = sc[ni][j] * 0.125f + mkv[ni];
                sc[ni][j] = v;
                tm = fmaxf(tm, v);
            }
            tmax[j] = tm;
        }
        // row-max butterfly across the 16-lane half holding this row
#pragma unroll
        for (int xm = 1; xm <= 8; xm <<= 1)
#pragma unroll
            for (int j = 0; j < 8; ++j)
                tmax[j] = fmaxf(tmax[j], __shfl_xor(tmax[j], xm));

        float scl[8];
#pragma unroll
        for (int j = 0; j < 8; ++j) {
            const float mn = fmaxf(mrow[j], tmax[j]);
            scl[j]  = __expf(mrow[j] - mn);
            mrow[j] = mn;
        }

        float rsum[8];
#pragma unroll
        for (int j = 0; j < 8; ++j) {
            float rs = 0.f;
#pragma unroll
            for (int ni = 0; ni < 4; ++ni) {
                float pv = __expf(sc[ni][j] - mrow[j]);
                sc[ni][j] = pv;
                rs += pv;
            }
            rsum[j] = rs;
        }
#pragma unroll
        for (int xm = 1; xm <= 8; xm <<= 1)
#pragma unroll
            for (int j = 0; j < 8; ++j)
                rsum[j] += __shfl_xor(rsum[j], xm);

#pragma unroll
        for (int j = 0; j < 8; ++j)
            lrow[j] = lrow[j] * scl[j] + rsum[j];
#pragma unroll
        for (int nd = 0; nd < 4; ++nd)
#pragma unroll
            for (int j = 0; j < 8; ++j)
                o[nd][j] *= scl[j];

        // ---- stage P (C-layout) -> LDS -> A-layout ----
#pragma unroll
        for (int ni = 0; ni < 4; ++ni)
#pragma unroll
            for (int j = 0; j < 8; ++j)
                pb[wave][j + laneHi * 8][ni * 16 + laneM] = (_Float16)sc[ni][j];

        asm volatile("s_wait_dscnt 0" ::: "memory");   // per-wave LDS RAW fence

        // ---- ctx += P @ V  (K-dim = 64 keys => two 32-steps) ----
#pragma unroll
        for (int step = 0; step < 2; ++step) {
            const _Float16* pp = &pb[wave][laneM][step * 32 + laneHi * 8];
            v16h pa = mk16(*(const v8h*)pp, *(const v8h*)(pp + 16));
#pragma unroll
            for (int nd = 0; nd < 4; ++nd) {
                const int n = nd * 16 + laneM;          // head-dim column
                const _Float16* pv = Vt + vbase + (size_t)n * NS
                                   + kt + step * 32 + laneHi * 8;
                v16h vb = mk16(ld8(pv), ld8(pv + 16));
                o[nd] = wmma16(pa, vb, o[nd]);
            }
        }
        // LDS ops are in-order per wave: next iteration's stores can't pass loads.
    }

    // ---- normalize + store fp32 output [B,S,H] ----
#pragma unroll
    for (int j = 0; j < 8; ++j) {
        const float inv = 1.0f / lrow[j];
        const int m = q0 + j + laneHi * 8;
#pragma unroll
        for (int nd = 0; nd < 4; ++nd) {
            const int n = nd * 16 + laneM;
            out[((size_t)b * NS + m) * NH_ + h * DH_ + n] = o[nd][j] * inv;
        }
    }
}

// ---------------------------------------------------------------------------
// Host-side launcher
// ---------------------------------------------------------------------------
extern "C" void kernel_launch(void* const* d_in, const int* in_sizes, int n_in,
                              void* d_out, int out_size, void* d_ws, size_t ws_size,
                              hipStream_t stream) {
    const float* hidden = (const float*)d_in[0];
    const float* mask   = (const float*)d_in[1];
    const float* Wq     = (const float*)d_in[2];
    const float* bq     = (const float*)d_in[3];
    const float* Wk     = (const float*)d_in[4];
    const float* bk     = (const float*)d_in[5];
    const float* Wv     = (const float*)d_in[6];
    const float* bv     = (const float*)d_in[7];
    const int*   eidx   = (const int*)d_in[8];
    float* out = (float*)d_out;

    // workspace layout (bytes)
    const size_t XH_ELEMS = (size_t)NB * NS * NH_;          // 6,291,456
    const size_t W_ELEMS  = (size_t)NE * NH_ * NH_;         // 4,718,592 per proj
    const size_t QKV_ELEMS = XH_ELEMS;                      // per tensor

    char* ws = (char*)d_ws;
    size_t off = 0;
    _Float16* Xh   = (_Float16*)(ws + off); off += XH_ELEMS * 2;
    _Float16* Wall = (_Float16*)(ws + off); off += 3 * W_ELEMS * 2;   // [3][E][H][H]
    _Float16* Qh   = (_Float16*)(ws + off); off += QKV_ELEMS * 2;
    _Float16* Khp  = (_Float16*)(ws + off); off += QKV_ELEMS * 2;
    _Float16* Vt   = (_Float16*)(ws + off); off += QKV_ELEMS * 2;
    (void)ws_size; (void)in_sizes; (void)n_in; (void)out_size;

    // f32 -> f16 conversions
    {
        int n4 = (int)(XH_ELEMS / 4);
        cvt_f32_to_f16_4<<<(n4 + 255) / 256, 256, 0, stream>>>(
            (const float4*)hidden, (v4h*)Xh, n4);
        int w4 = (int)(W_ELEMS / 4);
        cvt_f32_to_f16_4<<<(w4 + 255) / 256, 256, 0, stream>>>(
            (const float4*)Wq, (v4h*)(Wall + 0 * W_ELEMS), w4);
        cvt_f32_to_f16_4<<<(w4 + 255) / 256, 256, 0, stream>>>(
            (const float4*)Wk, (v4h*)(Wall + 1 * W_ELEMS), w4);
        cvt_f32_to_f16_4<<<(w4 + 255) / 256, 256, 0, stream>>>(
            (const float4*)Wv, (v4h*)(Wall + 2 * W_ELEMS), w4);
    }

    // expert-gathered QKV projections
    {
        dim3 grid(NS / 128, NH_ / 128, NB * 3);   // (4, 6, 48)
        qkv_wmma_kernel<<<grid, 256, 0, stream>>>(
            Xh, Wall, bq, bk, bv, eidx, Qh, Khp, Vt);
    }

    // attention
    {
        dim3 grid(NS / 64, NB * NHEAD);           // (8, 192)
        attn_wmma_kernel<<<grid, 128, 0, stream>>>(Qh, Khp, Vt, mask, out);
    }
}
